// MultiSelfAttention_69810398429712
// MI455X (gfx1250) — matrix operand order repaired and checked by reading, not compiled, not run on hardware
//
#include <hip/hip_runtime.h>
#include <hip/hip_bf16.h>

typedef __bf16 bf16_t;
typedef __attribute__((ext_vector_type(4)))  __bf16 v4bf;
typedef __attribute__((ext_vector_type(8)))  __bf16 v8bf;
typedef __attribute__((ext_vector_type(16))) __bf16 v16bf;
typedef __attribute__((ext_vector_type(8)))  float  v8f;

#define D_MODEL 1024
#define NHEAD   16
#define HDIM    64
#define SEQ     2048
#define BATCH   2

// Join two 16-byte LDS chunks into one WMMA operand register block.
static __device__ __forceinline__ v16bf frag2(const bf16_t* p0, const bf16_t* p1) {
  v8bf a = *(const v8bf*)p0;
  v8bf b = *(const v8bf*)p1;
  return __builtin_shufflevector(a, b, 0,1,2,3,4,5,6,7,8,9,10,11,12,13,14,15);
}

// -------------------------------------------------------------------------
// f32 -> bf16 conversion, 4 elements per thread (b128 in, b64 out)
// -------------------------------------------------------------------------
__global__ void cvt_f32_bf16_v4(const float4* __restrict__ s, v4bf* __restrict__ d, int n4) {
  int i = blockIdx.x * blockDim.x + threadIdx.x;
  if (i < n4) {
    float4 f = s[i];
    v4bf o = {(__bf16)f.x, (__bf16)f.y, (__bf16)f.z, (__bf16)f.w};
    d[i] = o;
  }
}

// -------------------------------------------------------------------------
// Tiled GEMM: C[M x 1024] = A[M x 1024] * W^T  (W is [N x K] row-major).
// B tile kept in W's native [n][k] order so B-fragments are contiguous.
// 128 threads = 4 waves; 64x64 output tile; double-buffered LDS K-tiles.
// outMode 0: bf16 output to [B, H, S, 64] with optional fused RoPE + scale
// outMode 1: f32 output, row-major [M x N]
// outMode 2: bf16 output TRANSPOSED to [B, H, 64, S] (for V): per-lane v8bf
//            packs 8 consecutive sequence positions -> single b128 store
// -------------------------------------------------------------------------
#define GK_LD 40   // 32 data + 8 pad bf16 -> 80B rows (16B aligned)

__global__ __launch_bounds__(128)
void gemm_proj(const bf16_t* __restrict__ A, const bf16_t* __restrict__ W,
               void* __restrict__ outp, const int* __restrict__ posp,
               int outMode, int doRope, float outScale)
{
  __shared__ alignas(16) bf16_t As[2][64 * GK_LD];   // [m][k]
  __shared__ alignas(16) bf16_t Bs[2][64 * GK_LD];   // [n][k]

  const int tid  = threadIdx.x;
  const int lane = tid & 31;
  const int wid  = tid >> 5;     // wave 0..3 -> 16-row band
  const int hf   = lane >> 4;    // half-wave 0/1
  const int lm   = lane & 15;
  const int mblk = blockIdx.x * 64;
  const int nblk = blockIdx.y * 64;

  // copy coordinates: each thread moves 2 uint4 per matrix per K-step
  const int cr0 = tid >> 2;       // rows 0..31
  const int cr1 = cr0 + 32;       // rows 32..63
  const int cc  = (tid & 3) * 8;  // bf16 column within 32-wide tile

  const bf16_t* Abase = A + (size_t)mblk * D_MODEL;
  const bf16_t* Wbase = W + (size_t)nblk * D_MODEL;

  v8f zero = {0.f,0.f,0.f,0.f,0.f,0.f,0.f,0.f};
  v8f acc[4];
  #pragma unroll
  for (int t = 0; t < 4; ++t) acc[t] = zero;

  // prologue: stage K-tile 0 into buffer 0
  *(uint4*)&As[0][cr0 * GK_LD + cc] = *(const uint4*)&Abase[(size_t)cr0 * D_MODEL + cc];
  *(uint4*)&As[0][cr1 * GK_LD + cc] = *(const uint4*)&Abase[(size_t)cr1 * D_MODEL + cc];
  *(uint4*)&Bs[0][cr0 * GK_LD + cc] = *(const uint4*)&Wbase[(size_t)cr0 * D_MODEL + cc];
  *(uint4*)&Bs[0][cr1 * GK_LD + cc] = *(const uint4*)&Wbase[(size_t)cr1 * D_MODEL + cc];
  __syncthreads();

  auto compute = [&](int cur) {
    // group ALL fragment loads first, then batch the 4 WMMAs
    const bf16_t* ap = &As[cur][(wid * 16 + lm) * GK_LD];
    v16bf af = frag2(ap + 16 * hf, ap + 8 + 16 * hf);
    v16bf bfr[4];
    #pragma unroll
    for (int nt = 0; nt < 4; ++nt) {
      const bf16_t* bp = &Bs[cur][(nt * 16 + lm) * GK_LD + 16 * hf];
      bfr[nt] = frag2(bp, bp + 8);
    }
    #pragma unroll
    for (int nt = 0; nt < 4; ++nt)
      acc[nt] = __builtin_amdgcn_wmma_f32_16x16x32_bf16(
          false, af, false, bfr[nt], (short)0, acc[nt], false, false);
  };

  for (int kk = 0; kk < (D_MODEL / 32) - 1; ++kk) {
    const int cur = kk & 1, nxt = cur ^ 1;
    const int k0n = (kk + 1) * 32;
    // prefetch next K-tile into registers (overlaps with WMMA below)
    uint4 ra0 = *(const uint4*)&Abase[(size_t)cr0 * D_MODEL + k0n + cc];
    uint4 ra1 = *(const uint4*)&Abase[(size_t)cr1 * D_MODEL + k0n + cc];
    uint4 rb0 = *(const uint4*)&Wbase[(size_t)cr0 * D_MODEL + k0n + cc];
    uint4 rb1 = *(const uint4*)&Wbase[(size_t)cr1 * D_MODEL + k0n + cc];

    compute(cur);

    *(uint4*)&As[nxt][cr0 * GK_LD + cc] = ra0;
    *(uint4*)&As[nxt][cr1 * GK_LD + cc] = ra1;
    *(uint4*)&Bs[nxt][cr0 * GK_LD + cc] = rb0;
    *(uint4*)&Bs[nxt][cr1 * GK_LD + cc] = rb1;
    __syncthreads();
  }
  compute(((D_MODEL / 32) - 1) & 1);

  if (outMode == 1) {
    float* O = (float*)outp;
    #pragma unroll
    for (int nt = 0; nt < 4; ++nt)
      #pragma unroll
      for (int v = 0; v < 8; ++v) {
        int row = mblk + wid * 16 + v + 8 * hf;
        int col = nblk + nt * 16 + lm;
        O[(size_t)row * D_MODEL + col] = acc[nt][v] * outScale;
      }
  } else if (outMode == 2) {
    // V: transposed [B, H, 64(d), S]; 8 accumulator rows of a lane are 8
    // consecutive sequence positions -> one aligned b128 store per tile.
    bf16_t* O = (bf16_t*)outp;
    #pragma unroll
    for (int nt = 0; nt < 4; ++nt) {
      v8bf pk;
      #pragma unroll
      for (int v = 0; v < 8; ++v) pk[v] = (bf16_t)(acc[nt][v] * outScale);
      int row0 = mblk + wid * 16 + 8 * hf;   // s base; v spans s0..s0+7
      int bb   = row0 >> 11;
      int s0   = row0 & (SEQ - 1);
      int col  = nblk + nt * 16 + lm;        // global feature = h*64 + d
      int hh = col >> 6, d2 = col & (HDIM - 1);
      *(v8bf*)&O[((size_t)((bb * NHEAD + hh) * HDIM + d2)) * SEQ + s0] = pk;
    }
  } else {
    bf16_t* O = (bf16_t*)outp;
    #pragma unroll
    for (int nt = 0; nt < 4; ++nt)
      #pragma unroll
      for (int v = 0; v < 8; ++v) {
        int row = mblk + wid * 16 + v + 8 * hf;
        int bb  = row >> 11;            // row / SEQ
        int s   = row & (SEQ - 1);
        int col = nblk + nt * 16 + lm;
        float val = acc[nt][v];
        if (doRope) {
          // adjacent feature column lives in adjacent lane (C layout col=lane%16)
          float prt = __shfl_xor(val, 1, 32);
          int d = col & (HDIM - 1);
          float fpos = (float)posp[s];
          // inv_freq = 10000^(-(2i)/64) ; 2i = d&~1 ; log2(10000)=13.2877124
          float inv = exp2f(-(float)(d & ~1) * (13.287712379549449f / 64.0f));
          float ang = fpos * inv;
          float sn, cs;
          __sincosf(ang, &sn, &cs);
          val = (d & 1) ? (prt * sn + val * cs) : (val * cs - prt * sn);
        }
        val *= outScale;
        int hh = col >> 6, d2 = col & (HDIM - 1);
        O[(((size_t)(bb * NHEAD + hh) * SEQ) + s) * HDIM + d2] = (bf16_t)val;
      }
  }
}

// -------------------------------------------------------------------------
// Flash attention over one (b, h, 64-row q-block). Q pre-scaled by 1/8.
// Online softmax; causal lower-triangle blocks only; double-buffered K/V.
// K is [B,H,S,64] ([kk][d] tiles); V is pre-transposed [B,H,64,S] so both
// stage with plain b128 copies and all B-frags are contiguous.
// Output bf16 [B,S,1024].
// -------------------------------------------------------------------------
#define AT_LD 72   // 64 data + 8 pad bf16 -> 144B rows (16B aligned)

__global__ __launch_bounds__(128)
void attn_flash(const bf16_t* __restrict__ Qb, const bf16_t* __restrict__ Kb,
                const bf16_t* __restrict__ Vb, bf16_t* __restrict__ Ob)
{
  __shared__ alignas(16) bf16_t Qs[64 * AT_LD];      // [q][d]
  __shared__ alignas(16) bf16_t Ps[64 * AT_LD];      // [q][kk], wave-private bands
  __shared__ alignas(16) bf16_t Ks[2][64 * AT_LD];   // [kk][d]
  __shared__ alignas(16) bf16_t Vt[2][64 * AT_LD];   // [d][kk]

  const int tid  = threadIdx.x;
  const int lane = tid & 31;
  const int wid  = tid >> 5;
  const int hf   = lane >> 4;
  const int lm   = lane & 15;

  const int qb = blockIdx.x;
  const int h  = blockIdx.y;
  const int b  = blockIdx.z;
  const int q0 = qb * 64;

  const size_t headoff = (size_t)(b * NHEAD + h) * SEQ * HDIM;
  const bf16_t* Qh = Qb + headoff;
  const bf16_t* Kh = Kb + headoff;
  const bf16_t* Vh = Vb + headoff;   // [64(d)][SEQ]

  // copy coordinates: 4 uint4 chunks per thread per 64x64 bf16 tile
  const int kr  = tid >> 3;        // base row (0..15), rows kr+16*it
  const int kc  = (tid & 7) * 8;   // bf16 column

  // Load Q tile + stage K/V block 0 into buffer 0 (all straight b128 copies)
  #pragma unroll
  for (int it = 0; it < 4; ++it) {
    int r = kr + 16 * it;
    *(uint4*)&Qs[r * AT_LD + kc]    = *(const uint4*)&Qh[(size_t)(q0 + r) * HDIM + kc];
    *(uint4*)&Ks[0][r * AT_LD + kc] = *(const uint4*)&Kh[(size_t)r * HDIM + kc];
    *(uint4*)&Vt[0][r * AT_LD + kc] = *(const uint4*)&Vh[(size_t)r * SEQ + kc];
  }
  __syncthreads();

  // Hoisted Q A-fragments (head dim 64 = two K-steps of 32)
  const bf16_t* qp = &Qs[(wid * 16 + lm) * AT_LD];
  v16bf aq0 = frag2(qp + 16 * hf,      qp + 8 + 16 * hf);
  v16bf aq1 = frag2(qp + 32 + 16 * hf, qp + 40 + 16 * hf);

  float mrow[8], lrow[8];
  v8f zero = {0.f,0.f,0.f,0.f,0.f,0.f,0.f,0.f};
  v8f acc[4];
  #pragma unroll
  for (int v = 0; v < 8; ++v) { mrow[v] = -1e30f; lrow[v] = 0.f; }
  #pragma unroll
  for (int t = 0; t < 4; ++t) acc[t] = zero;

  for (int kblk = 0; kblk <= qb; ++kblk) {
    const int cur = kblk & 1;
    const int k0  = kblk * 64;
    const bool pre = (kblk < qb);

    // prefetch next K/V block into registers (overlaps with compute)
    uint4 rk[4], rv[4];
    if (pre) {
      const int k0n = k0 + 64;
      #pragma unroll
      for (int it = 0; it < 4; ++it) {
        int r = kr + 16 * it;
        rk[it] = *(const uint4*)&Kh[(size_t)(k0n + r) * HDIM + kc];
        rv[it] = *(const uint4*)&Vh[(size_t)r * SEQ + k0n + kc];
      }
    }

    // ---- scores S = Q * K^T : load all B-frags, then batch 8 WMMAs
    v16bf bk[4][2];
    #pragma unroll
    for (int nt = 0; nt < 4; ++nt) {
      const bf16_t* kp = &Ks[cur][(nt * 16 + lm) * AT_LD + 16 * hf];
      bk[nt][0] = frag2(kp,      kp + 8);
      bk[nt][1] = frag2(kp + 32, kp + 40);
    }
    v8f sc[4];
    #pragma unroll
    for (int nt = 0; nt < 4; ++nt) {
      sc[nt] = zero;
      sc[nt] = __builtin_amdgcn_wmma_f32_16x16x32_bf16(false, aq0, false, bk[nt][0], (short)0, sc[nt], false, false);
      sc[nt] = __builtin_amdgcn_wmma_f32_16x16x32_bf16(false, aq1, false, bk[nt][1], (short)0, sc[nt], false, false);
    }
    float s[4][8];
    #pragma unroll
    for (int nt = 0; nt < 4; ++nt)
      #pragma unroll
      for (int v = 0; v < 8; ++v) {
        int qg = q0 + wid * 16 + v + 8 * hf;
        int kg = k0 + nt * 16 + lm;
        s[nt][v] = (kg <= qg) ? sc[nt][v] : -1e30f;
      }

    // ---- online softmax (row reductions across each 16-lane half)
    #pragma unroll
    for (int v = 0; v < 8; ++v) {
      float mb = fmaxf(fmaxf(s[0][v], s[1][v]), fmaxf(s[2][v], s[3][v]));
      #pragma unroll
      for (int off = 1; off < 16; off <<= 1)
        mb = fmaxf(mb, __shfl_xor(mb, off, 32));
      float mnew  = fmaxf(mrow[v], mb);
      float alpha = __expf(mrow[v] - mnew);
      float p[4];
      float rs = 0.f;
      #pragma unroll
      for (int nt = 0; nt < 4; ++nt) { p[nt] = __expf(s[nt][v] - mnew); rs += p[nt]; }
      #pragma unroll
      for (int off = 1; off < 16; off <<= 1)
        rs += __shfl_xor(rs, off, 32);
      lrow[v] = lrow[v] * alpha + rs;
      mrow[v] = mnew;
      #pragma unroll
      for (int dt = 0; dt < 4; ++dt) acc[dt][v] = acc[dt][v] * alpha;
      // re-layout probs C-frag -> LDS (wave-private rows; intra-wave ordering)
      const int prow = wid * 16 + v + 8 * hf;
      #pragma unroll
      for (int nt = 0; nt < 4; ++nt)
        Ps[prow * AT_LD + nt * 16 + lm] = (bf16_t)p[nt];
    }

    // ---- O += P * V : load all frags, then batch 8 WMMAs
    const bf16_t* pp = &Ps[(wid * 16 + lm) * AT_LD];
    v16bf ap0 = frag2(pp + 16 * hf,      pp + 8 + 16 * hf);
    v16bf ap1 = frag2(pp + 32 + 16 * hf, pp + 40 + 16 * hf);
    v16bf bv[4][2];
    #pragma unroll
    for (int dt = 0; dt < 4; ++dt) {
      const bf16_t* vp = &Vt[cur][(dt * 16 + lm) * AT_LD + 16 * hf];
      bv[dt][0] = frag2(vp,      vp + 8);
      bv[dt][1] = frag2(vp + 32, vp + 40);
    }
    #pragma unroll
    for (int dt = 0; dt < 4; ++dt) {
      acc[dt] = __builtin_amdgcn_wmma_f32_16x16x32_bf16(false, ap0, false, bv[dt][0], (short)0, acc[dt], false, false);
      acc[dt] = __builtin_amdgcn_wmma_f32_16x16x32_bf16(false, ap1, false, bv[dt][1], (short)0, acc[dt], false, false);
    }

    // ---- commit prefetched K/V into the alternate buffer (b128 stores)
    if (pre) {
      const int nxt = cur ^ 1;
      #pragma unroll
      for (int it = 0; it < 4; ++it) {
        int r = kr + 16 * it;
        *(uint4*)&Ks[nxt][r * AT_LD + kc] = rk[it];
        *(uint4*)&Vt[nxt][r * AT_LD + kc] = rv[it];
      }
    }
    __syncthreads();
  }

  // ---- epilogue: normalize and store bf16 [B, S, 1024]
  #pragma unroll
  for (int v = 0; v < 8; ++v) {
    float invl = 1.0f / lrow[v];
    #pragma unroll
    for (int dt = 0; dt < 4; ++dt) {
      int qg = q0 + wid * 16 + v + 8 * hf;
      int dg = dt * 16 + lm;
      Ob[(size_t)(b * SEQ + qg) * D_MODEL + h * HDIM + dg] = (bf16_t)(acc[dt][v] * invl);
    }
  }
}

// -------------------------------------------------------------------------
extern "C" void kernel_launch(void* const* d_in, const int* in_sizes, int n_in,
                              void* d_out, int out_size, void* d_ws, size_t ws_size,
                              hipStream_t stream) {
  (void)in_sizes; (void)n_in; (void)out_size; (void)ws_size;
  const float* x   = (const float*)d_in[0];
  const int*   pos = (const int*)d_in[1];
  const float* wq  = (const float*)d_in[2];
  const float* wk  = (const float*)d_in[3];
  const float* wv  = (const float*)d_in[4];
  const float* wo  = (const float*)d_in[5];

  const size_t MD = (size_t)BATCH * SEQ * D_MODEL;   // 4M elems
  const size_t WD = (size_t)D_MODEL * D_MODEL;       // 1M elems

  bf16_t* xb  = (bf16_t*)d_ws;
  bf16_t* wqb = xb  + MD;
  bf16_t* wkb = wqb + WD;
  bf16_t* wvb = wkb + WD;
  bf16_t* wob = wvb + WD;
  bf16_t* Qb  = wob + WD;   // [B, H, S, 64]
  bf16_t* Kb  = Qb  + MD;   // [B, H, S, 64]
  bf16_t* Vb  = Kb  + MD;   // [B, H, 64, S]  (pre-transposed)
  bf16_t* Ab  = Vb  + MD;   // attention out, bf16 [B, S, 1024]

  const int MD4 = (int)(MD / 4), WD4 = (int)(WD / 4);
  cvt_f32_bf16_v4<<<(MD4 + 255) / 256, 256, 0, stream>>>((const float4*)x,  (v4bf*)xb,  MD4);
  cvt_f32_bf16_v4<<<(WD4 + 255) / 256, 256, 0, stream>>>((const float4*)wq, (v4bf*)wqb, WD4);
  cvt_f32_bf16_v4<<<(WD4 + 255) / 256, 256, 0, stream>>>((const float4*)wk, (v4bf*)wkb, WD4);
  cvt_f32_bf16_v4<<<(WD4 + 255) / 256, 256, 0, stream>>>((const float4*)wv, (v4bf*)wvb, WD4);
  cvt_f32_bf16_v4<<<(WD4 + 255) / 256, 256, 0, stream>>>((const float4*)wo, (v4bf*)wob, WD4);

  dim3 g(BATCH * SEQ / 64, D_MODEL / 64);   // (64, 16)
  // 1/sqrt(head_dim) = 0.125 folded (exactly) into Q
  gemm_proj<<<g, 128, 0, stream>>>(xb, wqb, Qb, pos, 0, 1, 0.125f);
  gemm_proj<<<g, 128, 0, stream>>>(xb, wkb, Kb, pos, 0, 1, 1.0f);
  gemm_proj<<<g, 128, 0, stream>>>(xb, wvb, Vb, pos, 2, 0, 1.0f);   // V transposed

  dim3 ga(SEQ / 64, NHEAD, BATCH);
  attn_flash<<<ga, 128, 0, stream>>>(Qb, Kb, Vb, Ab);

  gemm_proj<<<g, 128, 0, stream>>>(Ab, wob, d_out, pos, 1, 0, 1.0f);
}
